// VRWKV_SpatialMix_54159537602695
// MI455X (gfx1250) — matrix operand order repaired and checked by reading, not compiled, not run on hardware
//
#include <hip/hip_runtime.h>
#include <hip/hip_bf16.h>
#include <math.h>

typedef __attribute__((ext_vector_type(16))) __bf16          v16bf;
typedef __attribute__((ext_vector_type(8)))  float           v8f;
typedef __attribute__((ext_vector_type(8)))  unsigned short  v8us;

static constexpr int   Hh     = 64;
static constexpr int   Ww     = 64;
static constexpr int   Cc     = 256;
static constexpr int   Tt     = Hh * Ww;          // 4096
static constexpr int   CHUNK  = 128;
static constexpr int   NCH    = Tt / CHUNK;       // 32
static constexpr float EPSF   = 1.1920929e-07f;
static constexpr float CLAMPV = 60.0f;

__device__ __forceinline__ unsigned short f32_bf16(float f) {
    unsigned int u = __float_as_uint(f);
    unsigned int r = (u + 0x7FFFu + ((u >> 16) & 1u)) >> 16;   // RNE
    return (unsigned short)r;
}

union FragBF {
    v16bf bf;
    v8us  h[2];
};

// ---------------------------------------------------------------- weights f32 -> bf16
__global__ __launch_bounds__(256) void cvt_bf16_kernel(const float* __restrict__ in,
                                                       unsigned short* __restrict__ out,
                                                       int n) {
    int i = blockIdx.x * 256 + threadIdx.x;
    if (i < n) out[i] = f32_bf16(in[i]);
}

// ---------------------------------------------------------------- zigzag orders (static per scheme)
__global__ __launch_bounds__(256) void zigzag_kernel(int* __restrict__ o0,
                                                     int* __restrict__ o1) {
    int t = blockIdx.x * 256 + threadIdx.x;
    if (t >= Tt) return;
    // scheme 0: ('top-left','horizontal')
    int r  = t / Ww, cp = t % Ww;
    o0[t] = r * Ww + (((r & 1) == 0) ? cp : (Ww - 1 - cp));
    // scheme 1: ('bottom-right','vertical')
    int j = r, i = cp;                       // t' = j*W + i
    int colv = Ww - 1 - i;
    int rowv = (((colv & 1) == 0) ? (Hh - 1 - j) : j);
    o1[t] = rowv * Ww + colv;
}

// ---------------------------------------------------------------- depthwise 3x3 conv, (B,T,C) layout
__global__ __launch_bounds__(256) void dwconv_kernel(const float* __restrict__ x,
                                                     const float* __restrict__ wdw,
                                                     float* __restrict__ out) {
    int idx = blockIdx.x * 256 + threadIdx.x;      // over B*T*C
    int c = idx % Cc;
    int t = (idx / Cc) % Tt;
    int b = idx / (Cc * Tt);
    int y = t / Ww, xx = t % Ww;
    float s = 0.0f;
#pragma unroll
    for (int dy = -1; dy <= 1; ++dy) {
#pragma unroll
        for (int dx = -1; dx <= 1; ++dx) {
            int yy = y + dy, xs = xx + dx;
            if (yy >= 0 && yy < Hh && xs >= 0 && xs < Ww)
                s += x[((size_t)b * Tt + yy * Ww + xs) * Cc + c] *
                     wdw[c * 9 + (dy + 1) * 3 + (dx + 1)];
        }
    }
    out[idx] = s;
}

// ---------------------------------------------------------------- q_shift + zigzag gather + bf16 cvt
__global__ __launch_bounds__(256) void qshift_gather_kernel(const float* __restrict__ cur,
                                                            const int* __restrict__ order,
                                                            unsigned short* __restrict__ xs) {
    int idx = blockIdx.x * 256 + threadIdx.x;      // over B*T*C (t' = scan order)
    int c  = idx % Cc;
    int tp = (idx / Cc) % Tt;
    int b  = idx / (Cc * Tt);
    int s  = order[tp];
    int y = s / Ww, x = s % Ww;
    int q = c >> 6;                                // quarter (C/4 = 64)
    int sy = y, sx = x;
    bool ok;
    if      (q == 0) { sx = x - 1; ok = (x >= 1); }        // shift right
    else if (q == 1) { sx = x + 1; ok = (x + 1 < Ww); }    // shift left
    else if (q == 2) { sy = y - 1; ok = (y >= 1); }        // shift down
    else             { sy = y + 1; ok = (y + 1 < Hh); }    // shift up
    float v = ok ? cur[((size_t)b * Tt + sy * Ww + sx) * Cc + c] : 0.0f;
    xs[(size_t)b * Tt * Cc + (size_t)tp * Cc + c] = f32_bf16(v);
}

// plain f32 -> bf16 copy of cur (for the final projection)
__global__ __launch_bounds__(256) void cvt_act_kernel(const float* __restrict__ cur,
                                                      unsigned short* __restrict__ xs,
                                                      int n) {
    int i = blockIdx.x * 256 + threadIdx.x;
    if (i < n) xs[i] = f32_bf16(cur[i]);
}

// ---------------------------------------------------------------- fused K/V/R WMMA GEMM
// Y[t,o] = sum_c X[t,c]*W[o,c]  =>  A = X tile (MxK), B[k,n] = W[n,k] (rows of W are B columns)
// epilogue: EK = exp(clip(k)), V = v, SR = sigmoid(r)
__global__ __launch_bounds__(256) void gemm_kvr_kernel(const unsigned short* __restrict__ X,
                                                       const unsigned short* __restrict__ Wk,
                                                       const unsigned short* __restrict__ Wv,
                                                       const unsigned short* __restrict__ Wr,
                                                       float* __restrict__ EK,
                                                       float* __restrict__ V,
                                                       float* __restrict__ SR) {
    const int lane = threadIdx.x & 31;
    const int wave = threadIdx.x >> 5;
    const int nTilesN = Cc / 16;                         // 16
    const int tile = blockIdx.x * 8 + wave;
    const int tm = tile / nTilesN;
    const int tn = tile % nTilesN;

    const int lo = lane & 15;
    const int hi = lane >> 4;

    const unsigned short* arow = X  + (size_t)(tm * 16 + lo) * Cc + hi * 16;
    const size_t wbase         =      (size_t)(tn * 16 + lo) * Cc + hi * 16;
    const unsigned short* bkr  = Wk + wbase;
    const unsigned short* bvr  = Wv + wbase;
    const unsigned short* brr  = Wr + wbase;

    v8f ck = {}, cv = {}, cr = {};
#pragma unroll
    for (int kb = 0; kb < Cc; kb += 32) {
        FragBF a, bk, bv, br;
        a.h[0]  = *(const v8us*)(arow + kb);
        a.h[1]  = *(const v8us*)(arow + kb + 8);
        bk.h[0] = *(const v8us*)(bkr + kb);
        bk.h[1] = *(const v8us*)(bkr + kb + 8);
        bv.h[0] = *(const v8us*)(bvr + kb);
        bv.h[1] = *(const v8us*)(bvr + kb + 8);
        br.h[0] = *(const v8us*)(brr + kb);
        br.h[1] = *(const v8us*)(brr + kb + 8);
        ck = __builtin_amdgcn_wmma_f32_16x16x32_bf16(false, a.bf, false, bk.bf,
                                                     (short)0, ck, false, false);
        cv = __builtin_amdgcn_wmma_f32_16x16x32_bf16(false, a.bf, false, bv.bf,
                                                     (short)0, cv, false, false);
        cr = __builtin_amdgcn_wmma_f32_16x16x32_bf16(false, a.bf, false, br.bf,
                                                     (short)0, cr, false, false);
    }
#pragma unroll
    for (int i = 0; i < 8; ++i) {
        int row = tm * 16 + hi * 8 + i;
        int col = tn * 16 + lo;
        size_t o = (size_t)row * Cc + col;
        float kk = ck[i];
        EK[o] = __expf(fminf(fmaxf(kk, -CLAMPV), CLAMPV));
        V[o]  = cv[i];
        SR[o] = 1.0f / (1.0f + __expf(-cr[i]));
    }
}

// single-output WMMA GEMM for the final projection (writes f32 to d_out)
__global__ __launch_bounds__(256) void gemm_out_kernel(const unsigned short* __restrict__ X,
                                                       const unsigned short* __restrict__ W,
                                                       float* __restrict__ Y) {
    const int lane = threadIdx.x & 31;
    const int wave = threadIdx.x >> 5;
    const int nTilesN = Cc / 16;
    const int tile = blockIdx.x * 8 + wave;
    const int tm = tile / nTilesN;
    const int tn = tile % nTilesN;

    const int lo = lane & 15;
    const int hi = lane >> 4;

    const unsigned short* arow = X + (size_t)(tm * 16 + lo) * Cc + hi * 16;
    const unsigned short* brow = W + (size_t)(tn * 16 + lo) * Cc + hi * 16;

    v8f acc = {};
#pragma unroll
    for (int kb = 0; kb < Cc; kb += 32) {
        FragBF a, b;
        a.h[0] = *(const v8us*)(arow + kb);
        a.h[1] = *(const v8us*)(arow + kb + 8);
        b.h[0] = *(const v8us*)(brow + kb);
        b.h[1] = *(const v8us*)(brow + kb + 8);
        acc = __builtin_amdgcn_wmma_f32_16x16x32_bf16(false, a.bf, false, b.bf,
                                                      (short)0, acc, false, false);
    }
#pragma unroll
    for (int i = 0; i < 8; ++i) {
        int row = tm * 16 + hi * 8 + i;
        int col = tn * 16 + lo;
        Y[(size_t)row * Cc + col] = acc[i];
    }
}

// ---------------------------------------------------------------- WKV chunked scan
// state: num_t = d*num_{t-1} + ek_t*v_t ; den_t = d*den_{t-1} + ek_t
// y_t = sr_t * (num_{t-1} + first*ek_t*v_t) / (den_{t-1} + first*ek_t + eps)
__global__ __launch_bounds__(256) void scanA_kernel(const float* __restrict__ EK,
                                                    const float* __restrict__ V,
                                                    const float* __restrict__ sd, int j,
                                                    float* __restrict__ chN,
                                                    float* __restrict__ chD) {
    int idx = blockIdx.x * 256 + threadIdx.x;      // over B*NCH*C
    int c  = idx % Cc;
    int ch = (idx / Cc) % NCH;
    int b  = idx / (Cc * NCH);
    float ld = fminf(fmaxf(-sd[j * Cc + c] / (float)Tt, -CLAMPV), CLAMPV);
    float d  = __expf(ld);
    float an = 0.0f, ad = 0.0f;
    size_t base = ((size_t)b * Tt + ch * CHUNK) * Cc + c;
    for (int t = 0; t < CHUNK; ++t) {
        float ek = EK[base], vv = V[base];
        an = d * an + ek * vv;
        ad = d * ad + ek;
        base += Cc;
    }
    size_t o = ((size_t)b * NCH + ch) * Cc + c;
    chN[o] = an;
    chD[o] = ad;
}

__global__ __launch_bounds__(256) void scanB_kernel(const float* __restrict__ chN,
                                                    const float* __restrict__ chD,
                                                    const float* __restrict__ sd, int j,
                                                    float* __restrict__ caN,
                                                    float* __restrict__ caD) {
    int idx = blockIdx.x * 256 + threadIdx.x;      // over B*C
    int c = idx % Cc;
    int b = idx / Cc;
    float ld = fminf(fmaxf(-sd[j * Cc + c] / (float)Tt, -CLAMPV), CLAMPV);
    float dL = __expf(ld * (float)CHUNK);
    float n = 0.0f, dd = 0.0f;
    for (int ch = 0; ch < NCH; ++ch) {
        size_t o = ((size_t)b * NCH + ch) * Cc + c;
        caN[o] = n;
        caD[o] = dd;
        n  = dL * n  + chN[o];
        dd = dL * dd + chD[o];
    }
}

__global__ __launch_bounds__(256) void scanC_kernel(const float* __restrict__ EK,
                                                    const float* __restrict__ V,
                                                    const float* __restrict__ SR,
                                                    const float* __restrict__ sd,
                                                    const float* __restrict__ sf, int j,
                                                    const float* __restrict__ caN,
                                                    const float* __restrict__ caD,
                                                    const int* __restrict__ order,
                                                    float* __restrict__ cur) {
    int idx = blockIdx.x * 256 + threadIdx.x;      // over B*NCH*C
    int c  = idx % Cc;
    int ch = (idx / Cc) % NCH;
    int b  = idx / (Cc * NCH);
    float ld = fminf(fmaxf(-sd[j * Cc + c] / (float)Tt, -CLAMPV), CLAMPV);
    float d  = __expf(ld);
    float first = __expf(fminf(fmaxf(sf[j * Cc + c] / (float)Tt, -CLAMPV), CLAMPV));
    size_t co = ((size_t)b * NCH + ch) * Cc + c;
    float num = caN[co], den = caD[co];
    int t0 = ch * CHUNK;
    size_t base = ((size_t)b * Tt + t0) * Cc + c;
    for (int t = 0; t < CHUNK; ++t) {
        float ek = EK[base], vv = V[base], sr = SR[base];
        float a = first * ek;
        float y = sr * (num + a * vv) / (den + a + EPSF);
        int s = order[t0 + t];                     // inverse permutation: cur[order[t']] = y[t']
        cur[((size_t)b * Tt + s) * Cc + c] = y;
        num = d * num + ek * vv;
        den = d * den + ek;
        base += Cc;
    }
}

// ================================================================ launch
extern "C" void kernel_launch(void* const* d_in, const int* in_sizes, int n_in,
                              void* d_out, int out_size, void* d_ws, size_t ws_size,
                              hipStream_t stream) {
    const float* x    = (const float*)d_in[0];
    const float* wdw  = (const float*)d_in[1];
    const float* wkey = (const float*)d_in[2];
    const float* wval = (const float*)d_in[3];
    const float* wrec = (const float*)d_in[4];
    const float* wout = (const float*)d_in[5];
    const float* sd   = (const float*)d_in[6];    // (2,C)
    const float* sf   = (const float*)d_in[7];    // (2,C)
    float* out = (float*)d_out;

    const int B  = in_sizes[0] / (Tt * Cc);
    const int BT = B * Tt;

    // -------- workspace carve-out (256B aligned)
    char* ws = (char*)d_ws;
    size_t off = 0;
    auto alloc = [&](size_t bytes) -> char* {
        size_t p = (off + 255) & ~size_t(255);
        off = p + bytes;
        return ws + p;
    };
    unsigned short* WkB = (unsigned short*)alloc((size_t)Cc * Cc * 2);
    unsigned short* WvB = (unsigned short*)alloc((size_t)Cc * Cc * 2);
    unsigned short* WrB = (unsigned short*)alloc((size_t)Cc * Cc * 2);
    unsigned short* WoB = (unsigned short*)alloc((size_t)Cc * Cc * 2);
    int* order0 = (int*)alloc((size_t)Tt * 4);
    int* order1 = (int*)alloc((size_t)Tt * 4);
    float* curF = (float*)alloc((size_t)BT * Cc * 4);
    unsigned short* xsB = (unsigned short*)alloc((size_t)BT * Cc * 2);
    float* EK = (float*)alloc((size_t)BT * Cc * 4);
    float* VV = (float*)alloc((size_t)BT * Cc * 4);
    float* SR = (float*)alloc((size_t)BT * Cc * 4);
    float* chN = (float*)alloc((size_t)B * NCH * Cc * 4);
    float* chD = (float*)alloc((size_t)B * NCH * Cc * 4);
    float* caN = (float*)alloc((size_t)B * NCH * Cc * 4);
    float* caD = (float*)alloc((size_t)B * NCH * Cc * 4);
    (void)ws_size; (void)n_in; (void)out_size;

    const int nElem   = BT * Cc;
    const int gElem   = nElem / 256;              // exact (BT*C multiple of 256)
    const int gTiles  = (BT / 16) * (Cc / 16) / 8;
    const int gScanAC = B * NCH * Cc / 256;
    const int gScanB  = (B * Cc + 255) / 256;
    const int nW      = Cc * Cc;
    const int gW      = (nW + 255) / 256;

    // weights -> bf16
    cvt_bf16_kernel<<<gW, 256, 0, stream>>>(wkey, WkB, nW);
    cvt_bf16_kernel<<<gW, 256, 0, stream>>>(wval, WvB, nW);
    cvt_bf16_kernel<<<gW, 256, 0, stream>>>(wrec, WrB, nW);
    cvt_bf16_kernel<<<gW, 256, 0, stream>>>(wout, WoB, nW);

    // static zigzag orders
    zigzag_kernel<<<(Tt + 255) / 256, 256, 0, stream>>>(order0, order1);

    // depthwise conv: x -> curF (spatial token order, (B,T,C))
    dwconv_kernel<<<gElem, 256, 0, stream>>>(x, wdw, curF);

    // two scan schemes
    for (int j = 0; j < 2; ++j) {
        const int* order = (j == 0) ? order0 : order1;
        qshift_gather_kernel<<<gElem, 256, 0, stream>>>(curF, order, xsB);
        gemm_kvr_kernel<<<gTiles, 256, 0, stream>>>(xsB, WkB, WvB, WrB, EK, VV, SR);
        scanA_kernel<<<gScanAC, 256, 0, stream>>>(EK, VV, sd, j, chN, chD);
        scanB_kernel<<<gScanB, 256, 0, stream>>>(chN, chD, sd, j, caN, caD);
        scanC_kernel<<<gScanAC, 256, 0, stream>>>(EK, VV, SR, sd, sf, j, caN, caD, order, curF);
    }

    // final projection: out = cur @ w_out^T
    cvt_act_kernel<<<gElem, 256, 0, stream>>>(curF, xsB, nElem);
    gemm_out_kernel<<<gTiles, 256, 0, stream>>>(xsB, WoB, out);
}
